// MinGRULayer_2594160247384
// MI455X (gfx1250) — matrix operand order repaired and checked by reading, not compiled, not run on hardware
//
#include <hip/hip_runtime.h>

// ---------------------------------------------------------------------------
// MinGRU: zs = sigmoid(xs@Wz^T + bz); hs = xs@Wh^T + bh;
//         s_t = (1-z_t)*s_{t-1} + z_t*h_t  (s_{-1}=0);  out = s@Wo^T + bo
// B=8 T=16384 D=H=O=256.  GEMMs via v_wmma_f32_16x16x32_bf16 (fp32 accum),
// 32 rows per block, double-buffered B fragments for latency hiding.
// Scan: 3-pass chunked parallel scan, fp32 carry, bf16 intermediates.
// ---------------------------------------------------------------------------

#define B_ 8
#define T_ 16384
#define D_ 256
#define H_ 256
#define O_ 256
#define N_ (B_ * T_)          // 131072 rows
#define CHUNK 256
#define NCHUNK (T_ / CHUNK)   // 64 chunks per sequence

typedef __attribute__((ext_vector_type(16))) __bf16 v16bf;
typedef __attribute__((ext_vector_type(8)))  __bf16 v8bf;
typedef __attribute__((ext_vector_type(8)))  float  v8f;

union ABfrag { v16bf v; v8bf h[2]; };

// bf16 conversions (round-to-nearest-even), bit-exact and backend-safe.
__device__ __forceinline__ __bf16 f2bf(float f) {
    unsigned u = __float_as_uint(f);
    u = (u + 0x7FFFu + ((u >> 16) & 1u)) >> 16;
    unsigned short s = (unsigned short)u;
    return __builtin_bit_cast(__bf16, s);
}
__device__ __forceinline__ float bf2f(__bf16 v) {
    unsigned short s = __builtin_bit_cast(unsigned short, v);
    return __uint_as_float(((unsigned)s) << 16);
}

__device__ __forceinline__ v8f wmma_bf16(const ABfrag& a, const ABfrag& b, v8f c) {
    return __builtin_amdgcn_wmma_f32_16x16x32_bf16(
        false, a.v, false, b.v, (short)0, c, false, false);
}

// ---------------------------------------------------------------------------
// Kernel 0: weights fp32 -> bf16 (H*D == O*H == 65536 each)
// ---------------------------------------------------------------------------
__global__ __launch_bounds__(256) void convert_weights(
    const float* __restrict__ Wz, const float* __restrict__ Wh,
    const float* __restrict__ Wo,
    __bf16* __restrict__ Wz_bf, __bf16* __restrict__ Wh_bf,
    __bf16* __restrict__ Wo_bf)
{
    int i = blockIdx.x * 256 + threadIdx.x;
    if (i < H_ * D_) {
        Wz_bf[i] = f2bf(Wz[i]);
        Wh_bf[i] = f2bf(Wh[i]);
        Wo_bf[i] = f2bf(Wo[i]);
    }
}

// ---------------------------------------------------------------------------
// Kernel 1: gates GEMM.  32 rows (2 M-tiles) per block, 4096 blocks, 256 thr.
// Wave w owns output columns [32w,32w+32) for BOTH Z and H across both
// M-tiles: 8 accumulators, 64 WMMAs/wave.  B fragments double-buffered so
// global loads for K-step k+32 are in flight while WMMAs consume step k.
// Epilogue writes a = 1 - sigmoid(z), b = sigmoid(z)*h as bf16.
// ---------------------------------------------------------------------------
__global__ __launch_bounds__(256) void gemm_gates(
    const float* __restrict__ xs,
    const __bf16* __restrict__ Wz_bf, const float* __restrict__ bz,
    const __bf16* __restrict__ Wh_bf, const float* __restrict__ bh,
    __bf16* __restrict__ a_buf, __bf16* __restrict__ b_buf)
{
    __shared__ __bf16 lds[32][D_ + 8];   // +8 bf16 pad kills bank conflicts

    const int  tid  = threadIdx.x;
    const long row0 = (long)blockIdx.x * 32;

    // ---- stage A tile (32 x 256 fp32 -> bf16) into LDS, coalesced ----
    {
        const int r  = tid >> 3;          // 0..31
        const int c0 = (tid & 7) << 5;    // 0,32,...,224
        const float* src = xs + (row0 + r) * (long)D_ + c0;
#pragma unroll
        for (int v = 0; v < 4; ++v) {
            v8bf o;
#pragma unroll
            for (int j = 0; j < 8; ++j) o[j] = f2bf(src[v * 8 + j]);
            *(v8bf*)&lds[r][c0 + v * 8] = o;
        }
    }
    __syncthreads();

    const int wave = tid >> 5;
    const int lane = tid & 31;
    const int col0 = wave * 32;        // this wave's 32 output columns
    const int lrow = lane & 15;
    const int lhi  = lane >> 4;

    // accumulators: [m-tile][n-tile]
    v8f zacc00 = {}, zacc01 = {}, zacc10 = {}, zacc11 = {};
    v8f hacc00 = {}, hacc01 = {}, hacc10 = {}, hacc11 = {};

    // per-lane weight base pointers (column = lane&15, K = 8*lhi+[0,8) and 16+8*lhi+[0,8))
    const __bf16* wz0 = Wz_bf + (col0 +      lrow) * D_ + lhi * 8;
    const __bf16* wz1 = Wz_bf + (col0 + 16 + lrow) * D_ + lhi * 8;
    const __bf16* wh0 = Wh_bf + (col0 +      lrow) * D_ + lhi * 8;
    const __bf16* wh1 = Wh_bf + (col0 + 16 + lrow) * D_ + lhi * 8;

    auto loadB = [&](ABfrag& z0, ABfrag& z1, ABfrag& h0, ABfrag& h1, int k0) {
        z0.h[0] = *(const v8bf*)(wz0 + k0); z0.h[1] = *(const v8bf*)(wz0 + k0 + 16);
        z1.h[0] = *(const v8bf*)(wz1 + k0); z1.h[1] = *(const v8bf*)(wz1 + k0 + 16);
        h0.h[0] = *(const v8bf*)(wh0 + k0); h0.h[1] = *(const v8bf*)(wh0 + k0 + 16);
        h1.h[0] = *(const v8bf*)(wh1 + k0); h1.h[1] = *(const v8bf*)(wh1 + k0 + 16);
    };
    auto loadA = [&](ABfrag& a0, ABfrag& a1, int k0) {
        a0.h[0] = *(const v8bf*)&lds[lrow][k0 + lhi * 8];
        a0.h[1] = *(const v8bf*)&lds[lrow][k0 + 16 + lhi * 8];
        a1.h[0] = *(const v8bf*)&lds[16 + lrow][k0 + lhi * 8];
        a1.h[1] = *(const v8bf*)&lds[16 + lrow][k0 + 16 + lhi * 8];
    };

    ABfrag Za0, Za1, Ha0, Ha1;   // buffer set A (even K-steps)
    ABfrag Zb0, Zb1, Hb0, Hb1;   // buffer set B (odd K-steps)
    loadB(Za0, Za1, Ha0, Ha1, 0);

#pragma unroll
    for (int k0 = 0; k0 < D_; k0 += 64) {
        // prefetch odd step while even step computes
        loadB(Zb0, Zb1, Hb0, Hb1, k0 + 32);
        ABfrag A0, A1;
        loadA(A0, A1, k0);
        zacc00 = wmma_bf16(A0, Za0, zacc00);
        zacc01 = wmma_bf16(A0, Za1, zacc01);
        zacc10 = wmma_bf16(A1, Za0, zacc10);
        zacc11 = wmma_bf16(A1, Za1, zacc11);
        hacc00 = wmma_bf16(A0, Ha0, hacc00);
        hacc01 = wmma_bf16(A0, Ha1, hacc01);
        hacc10 = wmma_bf16(A1, Ha0, hacc10);
        hacc11 = wmma_bf16(A1, Ha1, hacc11);

        // prefetch next even step while odd step computes
        if (k0 + 64 < D_) loadB(Za0, Za1, Ha0, Ha1, k0 + 64);
        loadA(A0, A1, k0 + 32);
        zacc00 = wmma_bf16(A0, Zb0, zacc00);
        zacc01 = wmma_bf16(A0, Zb1, zacc01);
        zacc10 = wmma_bf16(A1, Zb0, zacc10);
        zacc11 = wmma_bf16(A1, Zb1, zacc11);
        hacc00 = wmma_bf16(A0, Hb0, hacc00);
        hacc01 = wmma_bf16(A0, Hb1, hacc01);
        hacc10 = wmma_bf16(A1, Hb0, hacc10);
        hacc11 = wmma_bf16(A1, Hb1, hacc11);
    }

    const float bz0 = bz[col0 + lrow], bz1 = bz[col0 + 16 + lrow];
    const float bh0 = bh[col0 + lrow], bh1 = bh[col0 + 16 + lrow];

#pragma unroll
    for (int m = 0; m < 2; ++m) {
        const v8f& zA = m ? zacc10 : zacc00;
        const v8f& zB = m ? zacc11 : zacc01;
        const v8f& hA = m ? hacc10 : hacc00;
        const v8f& hB = m ? hacc11 : hacc01;
#pragma unroll
        for (int i = 0; i < 8; ++i) {
            const long row = row0 + m * 16 + i + lhi * 8;   // C layout
            const long o0  = row * H_ + col0 + lrow;
            const long o1  = o0 + 16;
            float z0 = 1.0f / (1.0f + __expf(-(zA[i] + bz0)));
            float h0 = hA[i] + bh0;
            float z1 = 1.0f / (1.0f + __expf(-(zB[i] + bz1)));
            float h1 = hB[i] + bh1;
            a_buf[o0] = f2bf(1.0f - z0);
            b_buf[o0] = f2bf(z0 * h0);
            a_buf[o1] = f2bf(1.0f - z1);
            b_buf[o1] = f2bf(z1 * h1);
        }
    }
}

// ---------------------------------------------------------------------------
// Kernel 2a: per-chunk reduction -> (prod a, local scan end).
// thread = h channel (coalesced 512B/step across the block).
// ---------------------------------------------------------------------------
__global__ __launch_bounds__(256) void scan_pass1(
    const __bf16* __restrict__ a_buf, const __bf16* __restrict__ b_buf,
    float* __restrict__ sumA, float* __restrict__ sumB)
{
    const int h = threadIdx.x;
    const int c = blockIdx.x;   // chunk
    const int b = blockIdx.y;   // batch
    long base = ((long)b * T_ + (long)c * CHUNK) * H_ + h;
    float A = 1.0f, S = 0.0f;
#pragma unroll 4
    for (int t = 0; t < CHUNK; ++t) {
        __builtin_prefetch(a_buf + base + 8 * H_, 0, 1);
        __builtin_prefetch(b_buf + base + 8 * H_, 0, 1);
        float av = bf2f(a_buf[base]);
        float bv = bf2f(b_buf[base]);
        A *= av;
        S = av * S + bv;
        base += H_;
    }
    const long s = ((long)b * NCHUNK + c) * H_ + h;
    sumA[s] = A;
    sumB[s] = S;
}

// ---------------------------------------------------------------------------
// Kernel 2b: sequential scan over chunk summaries -> chunk entry states.
// ---------------------------------------------------------------------------
__global__ __launch_bounds__(256) void scan_pass2(
    const float* __restrict__ sumA, const float* __restrict__ sumB,
    float* __restrict__ s0)
{
    const int h = threadIdx.x;
    const int b = blockIdx.x;
    float S = 0.0f;
    for (int c = 0; c < NCHUNK; ++c) {
        const long idx = ((long)b * NCHUNK + c) * H_ + h;
        s0[idx] = S;
        S = sumA[idx] * S + sumB[idx];
    }
}

// ---------------------------------------------------------------------------
// Kernel 2c: replay each chunk from its entry state; states overwrite a_buf
// in place (bf16) so GEMM-2 reads them directly.
// ---------------------------------------------------------------------------
__global__ __launch_bounds__(256) void scan_pass3(
    __bf16* __restrict__ a_buf, const __bf16* __restrict__ b_buf,
    const float* __restrict__ s0)
{
    const int h = threadIdx.x;
    const int c = blockIdx.x;
    const int b = blockIdx.y;
    long base = ((long)b * T_ + (long)c * CHUNK) * H_ + h;
    float S = s0[((long)b * NCHUNK + c) * H_ + h];
#pragma unroll 4
    for (int t = 0; t < CHUNK; ++t) {
        __builtin_prefetch(a_buf + base + 8 * H_, 0, 1);
        __builtin_prefetch(b_buf + base + 8 * H_, 0, 1);
        float av = bf2f(a_buf[base]);
        float bv = bf2f(b_buf[base]);
        S = av * S + bv;
        a_buf[base] = f2bf(S);
        base += H_;
    }
}

// ---------------------------------------------------------------------------
// Kernel 3: output GEMM: out = states(bf16) @ Wo^T + bo, fp32 out.
// Same tiling as kernel 1 (32 rows/block), double-buffered Wo fragments.
// ---------------------------------------------------------------------------
__global__ __launch_bounds__(256) void gemm_out(
    const __bf16* __restrict__ states, const __bf16* __restrict__ Wo_bf,
    const float* __restrict__ bo, float* __restrict__ out)
{
    __shared__ __bf16 lds[32][H_ + 8];

    const int  tid  = threadIdx.x;
    const long row0 = (long)blockIdx.x * 32;

    // ---- stage states tile (32 x 256 bf16) into LDS ----
    {
        const int r  = tid >> 3;
        const int c0 = (tid & 7) << 5;
        const __bf16* src = states + (row0 + r) * (long)H_ + c0;
#pragma unroll
        for (int v = 0; v < 4; ++v)
            *(v8bf*)&lds[r][c0 + v * 8] = *(const v8bf*)(src + v * 8);
    }
    __syncthreads();

    const int wave = tid >> 5;
    const int lane = tid & 31;
    const int col0 = wave * 32;
    const int lrow = lane & 15;
    const int lhi  = lane >> 4;

    v8f acc00 = {}, acc01 = {}, acc10 = {}, acc11 = {};

    const __bf16* wo0 = Wo_bf + (col0 +      lrow) * H_ + lhi * 8;
    const __bf16* wo1 = Wo_bf + (col0 + 16 + lrow) * H_ + lhi * 8;

    auto loadB = [&](ABfrag& b0, ABfrag& b1, int k0) {
        b0.h[0] = *(const v8bf*)(wo0 + k0); b0.h[1] = *(const v8bf*)(wo0 + k0 + 16);
        b1.h[0] = *(const v8bf*)(wo1 + k0); b1.h[1] = *(const v8bf*)(wo1 + k0 + 16);
    };
    auto loadA = [&](ABfrag& a0, ABfrag& a1, int k0) {
        a0.h[0] = *(const v8bf*)&lds[lrow][k0 + lhi * 8];
        a0.h[1] = *(const v8bf*)&lds[lrow][k0 + 16 + lhi * 8];
        a1.h[0] = *(const v8bf*)&lds[16 + lrow][k0 + lhi * 8];
        a1.h[1] = *(const v8bf*)&lds[16 + lrow][k0 + 16 + lhi * 8];
    };

    ABfrag Ba0, Ba1, Bb0, Bb1;
    loadB(Ba0, Ba1, 0);

#pragma unroll
    for (int k0 = 0; k0 < H_; k0 += 64) {
        loadB(Bb0, Bb1, k0 + 32);
        ABfrag A0, A1;
        loadA(A0, A1, k0);
        acc00 = wmma_bf16(A0, Ba0, acc00);
        acc01 = wmma_bf16(A0, Ba1, acc01);
        acc10 = wmma_bf16(A1, Ba0, acc10);
        acc11 = wmma_bf16(A1, Ba1, acc11);

        if (k0 + 64 < H_) loadB(Ba0, Ba1, k0 + 64);
        loadA(A0, A1, k0 + 32);
        acc00 = wmma_bf16(A0, Bb0, acc00);
        acc01 = wmma_bf16(A0, Bb1, acc01);
        acc10 = wmma_bf16(A1, Bb0, acc10);
        acc11 = wmma_bf16(A1, Bb1, acc11);
    }

    const float b0 = bo[col0 + lrow], b1 = bo[col0 + 16 + lrow];
#pragma unroll
    for (int m = 0; m < 2; ++m) {
        const v8f& aA = m ? acc10 : acc00;
        const v8f& aB = m ? acc11 : acc01;
#pragma unroll
        for (int i = 0; i < 8; ++i) {
            const long row = row0 + m * 16 + i + lhi * 8;
            out[row * (long)O_ + col0 + lrow]      = aA[i] + b0;
            out[row * (long)O_ + col0 + 16 + lrow] = aB[i] + b1;
        }
    }
}

// ---------------------------------------------------------------------------
// Launch.  Workspace layout (all 256B aligned):
//   Wz_bf/Wh_bf/Wo_bf : 3 * 128 KB
//   a_buf             : N*H bf16 = 64 MB   (becomes states after scan)
//   b_buf             : N*H bf16 = 64 MB
//   sumA/sumB/s0      : 3 * B*NCHUNK*H fp32 = 3 * 512 KB
// total ~130 MB.
// ---------------------------------------------------------------------------
extern "C" void kernel_launch(void* const* d_in, const int* in_sizes, int n_in,
                              void* d_out, int out_size, void* d_ws, size_t ws_size,
                              hipStream_t stream)
{
    (void)in_sizes; (void)n_in; (void)out_size; (void)ws_size;

    const float* xs = (const float*)d_in[0];
    const float* Wz = (const float*)d_in[1];
    const float* bz = (const float*)d_in[2];
    const float* Wh = (const float*)d_in[3];
    const float* bh = (const float*)d_in[4];
    const float* Wo = (const float*)d_in[5];
    const float* bo = (const float*)d_in[6];
    float* out = (float*)d_out;

    char* p = (char*)d_ws;
    auto carve = [&](size_t bytes) -> char* {
        char* r = p;
        p += (bytes + 255) & ~(size_t)255;
        return r;
    };
    __bf16* Wz_bf = (__bf16*)carve((size_t)H_ * D_ * 2);
    __bf16* Wh_bf = (__bf16*)carve((size_t)H_ * D_ * 2);
    __bf16* Wo_bf = (__bf16*)carve((size_t)O_ * H_ * 2);
    __bf16* a_buf = (__bf16*)carve((size_t)N_ * H_ * 2);
    __bf16* b_buf = (__bf16*)carve((size_t)N_ * H_ * 2);
    float*  sumA  = (float*)carve((size_t)B_ * NCHUNK * H_ * 4);
    float*  sumB  = (float*)carve((size_t)B_ * NCHUNK * H_ * 4);
    float*  s0    = (float*)carve((size_t)B_ * NCHUNK * H_ * 4);

    convert_weights<<<(H_ * D_ + 255) / 256, 256, 0, stream>>>(
        Wz, Wh, Wo, Wz_bf, Wh_bf, Wo_bf);

    gemm_gates<<<N_ / 32, 256, 0, stream>>>(
        xs, Wz_bf, bz, Wh_bf, bh, a_buf, b_buf);

    scan_pass1<<<dim3(NCHUNK, B_), 256, 0, stream>>>(a_buf, b_buf, sumA, sumB);
    scan_pass2<<<B_, 256, 0, stream>>>(sumA, sumB, s0);
    scan_pass3<<<dim3(NCHUNK, B_), 256, 0, stream>>>(a_buf, b_buf, s0);

    gemm_out<<<N_ / 32, 256, 0, stream>>>(a_buf, Wo_bf, bo, out);
}
